// approx_lenet5_59863254172206
// MI455X (gfx1250) — 8-bit matrix operand order repaired and checked by reading, compiled, not run on hardware
//
#include <hip/hip_runtime.h>
#include <stdint.h>

typedef __attribute__((ext_vector_type(8))) int v8i;
typedef __attribute__((ext_vector_type(4))) int v4i;

#define BATCH 4096

__device__ __forceinline__ float qscale_from_bits(const unsigned* mb) {
    return __uint_as_float(*mb) / 127.0f + 1e-8f;
}
__device__ __forceinline__ int quant_u8(float v, float s) {
    float q = fminf(fmaxf(rintf(v / s), -128.0f), 127.0f);
    return (int)q + 128;
}

__device__ __forceinline__ void async_cp16(void* lds, const void* g) {
#if __has_builtin(__builtin_amdgcn_global_load_async_to_lds_b128)
    __builtin_amdgcn_global_load_async_to_lds_b128(
        (__attribute__((address_space(1))) v4i*)g,
        (__attribute__((address_space(3))) v4i*)lds, 0, 0);
#else
    *(float4*)lds = *(const float4*)g;
#endif
}
__device__ __forceinline__ void wait_async0() {
#if __has_builtin(__builtin_amdgcn_s_wait_asynccnt)
    __builtin_amdgcn_s_wait_asynccnt(0);
#else
    asm volatile("s_wait_asynccnt 0" ::: "memory");
#endif
}
__device__ __forceinline__ void stage_table_256k(float* slut, const float* g, int tid, int nthr) {
    for (int i = tid; i < 16384; i += nthr)
        async_cp16(slut + (size_t)i * 4, g + (size_t)i * 4);
}


__global__ void k_init(unsigned* maxbits) {
    if (threadIdx.x < 5) maxbits[threadIdx.x] = 0u;
}

__global__ void k_maxabs(const float* __restrict__ p, int n, unsigned* out) {
    __shared__ float red[256];
    float m = 0.0f;
    for (int i = blockIdx.x * blockDim.x + threadIdx.x; i < n; i += gridDim.x * blockDim.x)
        m = fmaxf(m, fabsf(p[i]));
    red[threadIdx.x] = m;
    __syncthreads();
    for (int s = 128; s > 0; s >>= 1) {
        if (threadIdx.x < s) red[threadIdx.x] = fmaxf(red[threadIdx.x], red[threadIdx.x + s]);
        __syncthreads();
    }
    if (threadIdx.x == 0) atomicMax(out, __float_as_uint(red[0]));
}

__global__ void k_quantw(const float* __restrict__ w, int n,
                         unsigned char* __restrict__ qw, float* __restrict__ sw_out) {
    __shared__ float red[256];
    __shared__ float s_sw;
    float m = 0.0f;
    for (int i = threadIdx.x; i < n; i += 256) m = fmaxf(m, fabsf(w[i]));
    red[threadIdx.x] = m;
    __syncthreads();
    for (int s = 128; s > 0; s >>= 1) {
        if (threadIdx.x < s) red[threadIdx.x] = fmaxf(red[threadIdx.x], red[threadIdx.x + s]);
        __syncthreads();
    }
    if (threadIdx.x == 0) {
        float s = red[0] / 127.0f + 1e-8f;
        s_sw = s;
        *sw_out = s;
    }
    __syncthreads();
    float s = s_sw;
    for (int i = threadIdx.x; i < n; i += 256)
        qw[i] = (unsigned char)quant_u8(w[i], s);
}

__global__ void k_noise(const float* __restrict__ lut, float* __restrict__ noise) {
    int i = blockIdx.x * blockDim.x + threadIdx.x;
    if (i < 65536) {
        int a = i >> 8, b = i & 255;
        noise[i] = lut[i] - (float)((a - 128) * (b - 128));
    }
}

__global__ void k_conv1(const float* __restrict__ x, const float* __restrict__ lut,
                        const unsigned char* __restrict__ qw, const float* __restrict__ bias,
                        const unsigned* __restrict__ mb, const float* __restrict__ swp,
                        float* __restrict__ p1) {
    extern __shared__ unsigned char smem[];
    float* slut = (float*)smem;
    unsigned char* sqw = smem + 65536 * 4;
    stage_table_256k(slut, lut, threadIdx.x, blockDim.x);
    for (int i = threadIdx.x; i < 150; i += blockDim.x) sqw[i] = qw[i];
    wait_async0();
    __syncthreads();

    int idx = blockIdx.x * blockDim.x + threadIdx.x;
    if (idx >= BATCH * 144) return;
    int b = idx / 144, pos = idx % 144;
    int py = pos / 12, px = pos % 12;
    float s1 = qscale_from_bits(mb);
    float ss = s1 * (*swp);
    const float* xb = x + (size_t)b * 784;

    float best[6];
#pragma unroll
    for (int o = 0; o < 6; o++) best[o] = -1e30f;

    for (int dy = 0; dy < 2; dy++) {
        for (int dx = 0; dx < 2; dx++) {
            int oy = 2 * py + dy, ox = 2 * px + dx;
            int q[25];
#pragma unroll
            for (int r = 0; r < 5; r++)
#pragma unroll
                for (int c = 0; c < 5; c++)
                    q[r * 5 + c] = quant_u8(xb[(oy + r) * 28 + (ox + c)], s1);
#pragma unroll
            for (int o = 0; o < 6; o++) {
                float acc = 0.0f;
#pragma unroll
                for (int k = 0; k < 25; k++)
                    acc += slut[(q[k] << 8) + sqw[o * 25 + k]];
                best[o] = fmaxf(best[o], ss * acc + bias[o]);
            }
        }
    }
#pragma unroll
    for (int o = 0; o < 6; o++)
        p1[(((size_t)b * 6 + o) * 12 + py) * 12 + px] = fmaxf(best[o], 0.0f);
}

__global__ void k_conv2(const float* __restrict__ p1, const float* __restrict__ lut,
                        const unsigned char* __restrict__ qw, const float* __restrict__ bias,
                        const unsigned* __restrict__ mb, const float* __restrict__ swp,
                        float* __restrict__ p2) {
    extern __shared__ unsigned char smem[];
    float* slut = (float*)smem;
    unsigned char* sqw = smem + 65536 * 4;
    stage_table_256k(slut, lut, threadIdx.x, blockDim.x);
    for (int i = threadIdx.x; i < 2400; i += blockDim.x) sqw[i] = qw[i];
    wait_async0();
    __syncthreads();

    int idx = blockIdx.x * blockDim.x + threadIdx.x;
    if (idx >= BATCH * 16) return;
    int b = idx >> 4, pos = idx & 15;
    int py = pos >> 2, px = pos & 3;
    float s2 = qscale_from_bits(mb);
    float ss = s2 * (*swp);

    float best[16];
#pragma unroll
    for (int o = 0; o < 16; o++) best[o] = -1e30f;

    for (int dy = 0; dy < 2; dy++) {
        for (int dx = 0; dx < 2; dx++) {
            int oy = 2 * py + dy, ox = 2 * px + dx;
            float acc[16];
#pragma unroll
            for (int o = 0; o < 16; o++) acc[o] = 0.0f;
            for (int c = 0; c < 6; c++) {
                const float* pb = p1 + ((size_t)b * 6 + c) * 144;
                int q[25];
#pragma unroll
                for (int r = 0; r < 5; r++)
#pragma unroll
                    for (int cc = 0; cc < 5; cc++)
                        q[r * 5 + cc] = quant_u8(pb[(oy + r) * 12 + (ox + cc)], s2);
#pragma unroll
                for (int o = 0; o < 16; o++) {
                    const unsigned char* wr = sqw + o * 150 + c * 25;
                    float a2 = 0.0f;
#pragma unroll
                    for (int k = 0; k < 25; k++)
                        a2 += slut[(q[k] << 8) + wr[k]];
                    acc[o] += a2;
                }
            }
#pragma unroll
            for (int o = 0; o < 16; o++)
                best[o] = fmaxf(best[o], ss * acc[o] + bias[o]);
        }
    }
#pragma unroll
    for (int o = 0; o < 16; o++)
        p2[(size_t)b * 256 + o * 16 + py * 4 + px] = fmaxf(best[o], 0.0f);
}

template <int KPAD>
__global__ void k_fc_wmma(const float* __restrict__ in, const float* __restrict__ noise,
                          const unsigned char* __restrict__ qw, const float* __restrict__ bias,
                          const unsigned* __restrict__ mb, const float* __restrict__ swp,
                          float* __restrict__ out,
                          int K, int O, int Opad, int do_relu) {
    extern __shared__ unsigned char smem[];
    float* snoise = (float*)smem;
    unsigned char* sqa = smem + 262144;
    unsigned char* sqwT = sqa + 16 * KPAD;

    int tid = threadIdx.x, nthr = blockDim.x;
    int b0 = blockIdx.x * 16;
    float s = qscale_from_bits(mb);

    stage_table_256k(snoise, noise, tid, nthr);
    for (int i = tid; i < 16 * KPAD; i += nthr) {
        int m = i / KPAD, k = i & (KPAD - 1);
        sqa[i] = (k < K) ? (unsigned char)quant_u8(in[(size_t)(b0 + m) * K + k], s)
                         : (unsigned char)128;
    }
    for (int i = tid; i < KPAD * Opad; i += nthr) {
        int k = i / Opad, o = i - k * Opad;
        sqwT[i] = (o < O && k < K) ? qw[o * K + k] : (unsigned char)128;
    }
    wait_async0();
    __syncthreads();

    int wave = tid >> 5, lane = tid & 31;
    int o0 = wave * 16;
    int m_a = lane & 15;
    int khi = (lane >> 4) * 8;
    const int kb[8] = {0, 4, 16, 20, 32, 36, 48, 52};

    v8i acc = {0, 0, 0, 0, 0, 0, 0, 0};
#pragma unroll
    for (int ch = 0; ch < (KPAD >> 6); ch++) {
        v8i A, Bm;
#pragma unroll
        for (int v = 0; v < 8; v++) {
            int k0 = ch * 64 + kb[v] + khi;
            A[v] = (int)(*(const unsigned*)(sqa + m_a * KPAD + k0) ^ 0x80808080u);
            const unsigned char* pB = sqwT + k0 * Opad + o0 + m_a;
            unsigned wB = (unsigned)pB[0] | ((unsigned)pB[Opad] << 8) |
                          ((unsigned)pB[2 * Opad] << 16) | ((unsigned)pB[3 * Opad] << 24);
            Bm[v] = (int)(wB ^ 0x80808080u);
        }
        acc = __builtin_amdgcn_wmma_i32_16x16x64_iu8(true, A, true, Bm, acc, false, false);
    }

    int n = lane & 15;
    int mbase = (lane >> 4) * 8;
    int o = o0 + n;
    if (o < O) {
        float ns[8];
#pragma unroll
        for (int r = 0; r < 8; r++) ns[r] = 0.0f;
        for (int k = 0; k < K; k++) {
            int wv = sqwT[k * Opad + o];
#pragma unroll
            for (int r = 0; r < 8; r++)
                ns[r] += snoise[((int)sqa[(mbase + r) * KPAD + k] << 8) + wv];
        }
        float ssw = s * (*swp);
        float bia = bias[o];
#pragma unroll
        for (int r = 0; r < 8; r++) {
            float v = ssw * ((float)acc[r] + ns[r]) + bia;
            if (do_relu) v = fmaxf(v, 0.0f);
            out[(size_t)(b0 + mbase + r) * O + o] = v;
        }
    }
}


extern "C" void kernel_launch(void* const* d_in, const int* in_sizes, int n_in,
                              void* d_out, int out_size, void* d_ws, size_t ws_size,
                              hipStream_t stream) {
    (void)in_sizes; (void)n_in; (void)out_size; (void)ws_size;
    const float* x   = (const float*)d_in[0];
    const float* lut = (const float*)d_in[1];
    const float* c1w = (const float*)d_in[2];
    const float* c1b = (const float*)d_in[3];
    const float* c2w = (const float*)d_in[4];
    const float* c2b = (const float*)d_in[5];
    const float* f1w = (const float*)d_in[6];
    const float* f1b = (const float*)d_in[7];
    const float* f2w = (const float*)d_in[8];
    const float* f2b = (const float*)d_in[9];
    const float* f3w = (const float*)d_in[10];
    const float* f3b = (const float*)d_in[11];

    unsigned char* ws = (unsigned char*)d_ws;
    size_t off = 0;
    auto take = [&](size_t bytes) -> size_t {
        size_t o = off;
        off += (bytes + 255) & ~(size_t)255;
        return o;
    };
    unsigned* maxbits      = (unsigned*)(ws + take(5 * sizeof(unsigned)));
    float*    swv          = (float*)(ws + take(5 * sizeof(float)));
    float*    noise        = (float*)(ws + take((size_t)65536 * 4));
    unsigned char* qw1     = ws + take(6 * 25);
    unsigned char* qw2     = ws + take(16 * 150);
    unsigned char* qw3     = ws + take(120 * 256);
    unsigned char* qw4     = ws + take(84 * 120);
    unsigned char* qw5     = ws + take(10 * 84);
    float* p1 = (float*)(ws + take((size_t)BATCH * 6 * 144 * 4));
    float* p2 = (float*)(ws + take((size_t)BATCH * 256 * 4));
    float* f1 = (float*)(ws + take((size_t)BATCH * 120 * 4));
    float* f2 = (float*)(ws + take((size_t)BATCH * 84 * 4));

    k_init<<<1, 32, 0, stream>>>(maxbits);
    k_maxabs<<<1024, 256, 0, stream>>>(x, BATCH * 784, maxbits + 0);

    k_quantw<<<1, 256, 0, stream>>>(c1w, 6 * 25,    qw1, swv + 0);
    k_quantw<<<1, 256, 0, stream>>>(c2w, 16 * 150,  qw2, swv + 1);
    k_quantw<<<1, 256, 0, stream>>>(f1w, 120 * 256, qw3, swv + 2);
    k_quantw<<<1, 256, 0, stream>>>(f2w, 84 * 120,  qw4, swv + 3);
    k_quantw<<<1, 256, 0, stream>>>(f3w, 10 * 84,   qw5, swv + 4);

    k_noise<<<256, 256, 0, stream>>>(lut, noise);

    k_conv1<<<(BATCH * 144) / 256, 256, 262144 + 256, stream>>>(
        x, lut, qw1, c1b, maxbits + 0, swv + 0, p1);
    k_maxabs<<<1024, 256, 0, stream>>>(p1, BATCH * 6 * 144, maxbits + 1);

    k_conv2<<<(BATCH * 16) / 256, 256, 262144 + 2560, stream>>>(
        p1, lut, qw2, c2b, maxbits + 1, swv + 1, p2);
    k_maxabs<<<1024, 256, 0, stream>>>(p2, BATCH * 256, maxbits + 2);

    k_fc_wmma<256><<<BATCH / 16, 256, 262144 + 16 * 256 + 256 * 128, stream>>>(
        p2, noise, qw3, f1b, maxbits + 2, swv + 2, f1, 256, 120, 128, 1);
    k_maxabs<<<512, 256, 0, stream>>>(f1, BATCH * 120, maxbits + 3);

    k_fc_wmma<128><<<BATCH / 16, 192, 262144 + 16 * 128 + 128 * 96, stream>>>(
        f1, noise, qw4, f2b, maxbits + 3, swv + 3, f2, 120, 84, 96, 1);
    k_maxabs<<<512, 256, 0, stream>>>(f2, BATCH * 84, maxbits + 4);

    k_fc_wmma<128><<<BATCH / 16, 32, 262144 + 16 * 128 + 128 * 16, stream>>>(
        f2, noise, qw5, f3b, maxbits + 4, swv + 4, (float*)d_out, 84, 10, 16, 0);
}
